// SNet_fgn_ver2_39032662786627
// MI455X (gfx1250) — matrix operand order repaired and checked by reading, not compiled, not run on hardware
//
#include <hip/hip_runtime.h>
#include <stdint.h>

typedef __attribute__((ext_vector_type(16))) __bf16 v16bf;
typedef __attribute__((ext_vector_type(8)))  float  v8f;

#define BB 8
#define HH 256
#define WW 256
#define PIX (BB*HH*WW)

__device__ __forceinline__ unsigned short f2bf(float f) {
  union { float f; unsigned u; } c; c.f = f;
  unsigned r = ((c.u >> 16) & 1u) + 0x7fffu;
  return (unsigned short)((c.u + r) >> 16);
}
__device__ __forceinline__ float bf2f(unsigned short h) {
  union { unsigned u; float f; } c; c.u = ((unsigned)h) << 16;
  return c.f;
}

// ---------------------------------------------------------------------------
// Pack f32 OIHW (Cout,64,3,3) weights into bf16 [CoutPad][576], k = r*64+ci
// (r = dy*3+dx slowest) so every 32-wide K chunk is one tap x 32 channels.
// ---------------------------------------------------------------------------
__global__ void prep_w2(const float* __restrict__ w, unsigned short* __restrict__ w2,
                        int Cout, int CoutPad) {
  int i = blockIdx.x * 256 + threadIdx.x;
  if (i >= CoutPad * 576) return;
  int co = i / 576, k = i % 576;
  int r = k >> 6, ci = k & 63;
  float v = (co < Cout) ? w[(co * 64 + ci) * 9 + r] : 0.f;
  w2[i] = f2bf(v);
}

// ---------------------------------------------------------------------------
// Direct conv for small channel counts (conv_g branch), f32 channels-last.
// act: 0 none, 1 relu, 2 sigmoid.  Output written at p*outStride+outOff+co.
// ---------------------------------------------------------------------------
__global__ void conv_small(const float* __restrict__ in, const float* __restrict__ w,
                           const float* __restrict__ bias,
                           const float* __restrict__ bng, const float* __restrict__ bnb,
                           float* __restrict__ out,
                           int Cin, int Cout, int inStride, int outStride, int outOff, int act) {
  int idx = blockIdx.x * 256 + threadIdx.x;
  if (idx >= PIX * Cout) return;
  int co = idx % Cout; int p = idx / Cout;
  int x = p % WW; int y = (p / WW) % HH; int b = p / (WW * HH);
  float acc = bias[co];
  for (int ci = 0; ci < Cin; ++ci)
    for (int dy = 0; dy < 3; ++dy) {
      int gy = y + dy - 1; if ((unsigned)gy >= HH) continue;
      for (int dx = 0; dx < 3; ++dx) {
        int gx = x + dx - 1; if ((unsigned)gx >= WW) continue;
        acc += in[(size_t)((b * HH + gy) * WW + gx) * inStride + ci] *
               w[((co * Cin + ci) * 3 + dy) * 3 + dx];
      }
    }
  if (bng) acc = acc * bng[co] + bnb[co];
  if (act == 1) acc = fmaxf(acc, 0.f);
  else if (act == 2) acc = 1.f / (1.f + __expf(-acc));
  out[(size_t)p * outStride + outOff + co] = acc;
}

// ---------------------------------------------------------------------------
// Cin (1 or 2, f32) -> 64 channels bf16 channels-last, bias [+BN] + ReLU.
// One thread computes 8 couts for one pixel -> packed 16B store.
// ---------------------------------------------------------------------------
__global__ void conv_in64(const float* __restrict__ in, const float* __restrict__ w,
                          const float* __restrict__ bias,
                          const float* __restrict__ bng, const float* __restrict__ bnb,
                          unsigned short* __restrict__ out, int Cin, int inStride) {
  int idx = blockIdx.x * 256 + threadIdx.x;
  int grp = idx & 7; int p = idx >> 3;
  if (p >= PIX) return;
  int x = p % WW; int y = (p / WW) % HH; int b = p / (WW * HH);
  float v[18];
  for (int ci = 0; ci < Cin; ++ci)
    for (int dy = 0; dy < 3; ++dy)
      for (int dx = 0; dx < 3; ++dx) {
        int gy = y + dy - 1, gx = x + dx - 1;
        v[ci * 9 + dy * 3 + dx] = ((unsigned)gy < HH && (unsigned)gx < WW)
            ? in[(size_t)((b * HH + gy) * WW + gx) * inStride + ci] : 0.f;
      }
  unsigned short o[8];
  for (int j = 0; j < 8; ++j) {
    int co = grp * 8 + j;
    float acc = bias[co];
    for (int ci = 0; ci < Cin; ++ci)
      for (int r = 0; r < 9; ++r) acc += v[ci * 9 + r] * w[(co * Cin + ci) * 9 + r];
    if (bng) acc = acc * bng[co] + bnb[co];
    acc = fmaxf(acc, 0.f);
    o[j] = f2bf(acc);
  }
  *(uint4*)&out[(size_t)p * 64 + grp * 8] = *(uint4*)o;
}

// ---------------------------------------------------------------------------
// Main implicit-GEMM 3x3 conv, Cin=64, via v_wmma_f32_16x16x32_bf16.
// in : bf16 [B,H,W,64] channels-last; w2: bf16 [CoutPad][576] (k=r*64+ci)
// out: bf16 [B,H,W,CoutPad]; grid = (W/64, H, B*cgroups), block = 256.
// LDS staging uses gfx1250 GLOBAL_LOAD_ASYNC_TO_LDS_B128 (ASYNCcnt) for the
// in-bounds body; halo is zero-filled with plain ds stores.
// ---------------------------------------------------------------------------
__global__ __launch_bounds__(256)
void conv64_wmma(const unsigned short* __restrict__ in,
                 const unsigned short* __restrict__ w2,
                 const float* __restrict__ bias,
                 const float* __restrict__ bng, const float* __restrict__ bnb,
                 unsigned short* __restrict__ out,
                 int Cout, int CoutPad, int cgroups, int relu) {
  __shared__ unsigned short lds[3 * 66 * 72];   // [dy][xi<66][ci pad 72], 28512 B
  int tid = threadIdx.x;
  int x0 = blockIdx.x * 64;
  int y  = blockIdx.y;
  int coBase = (blockIdx.z % cgroups) * 64;
  int b = blockIdx.z / cgroups;

  // Stage 3 rows x 66 px x 64 ch of bf16 input into LDS (zero halo).
  for (int c = tid; c < 3 * 66 * 8; c += 256) {
    int part = c & 7; int dyxi = c >> 3;
    int dy = dyxi / 66, xi = dyxi - dy * 66;
    int gy = y + dy - 1, gx = x0 - 1 + xi;
    unsigned ldsOff = (unsigned)((dyxi * 72 + part * 8) * 2);      // LDS byte addr
    if ((unsigned)gy < HH && (unsigned)gx < WW) {
      unsigned goff = (unsigned)(((size_t)((b * HH + gy) * WW + gx) * 64 + part * 8) * 2);
      // async DMA: global (SADDR base + 32-bit byte offset) -> LDS[ldsOff]
      asm volatile("global_load_async_to_lds_b128 %0, %1, %2"
                   :: "v"(ldsOff), "v"(goff), "s"(in) : "memory");
    } else {
      uint4 z = {0u, 0u, 0u, 0u};
      *(uint4*)((char*)lds + ldsOff) = z;
    }
  }
  asm volatile("s_wait_asynccnt 0x0" ::: "memory");
  __syncthreads();

  int wave = tid >> 5, lane = tid & 31;
  int m  = wave & 3;            // cout 16-tile within 64-cout group
  int n0 = (wave >> 2) * 2;     // first of two pixel 16-tiles
  int nlo = lane & 15, h = lane >> 4;

  int mTiles = (CoutPad - coBase) >> 4; if (mTiles > 4) mTiles = 4;
  if (m >= mTiles) return;

  v8f acc0 = {}; v8f acc1 = {};
  const unsigned short* wrow = w2 + (size_t)(coBase + m * 16 + nlo) * 576;

  // 9 taps x 2 ci-halves; bounded unroll keeps VGPRs under 256 so >=2 blocks
  // (16 waves) stay resident per WGP.
  #pragma unroll 3
  for (int r = 0; r < 9; ++r) {
    int dy = r / 3, dx = r - dy * 3;
    #pragma unroll
    for (int half = 0; half < 2; ++half) {
      int kstep = r * 2 + half;
      int ci0 = half << 5;
      // A fragment: row M=nlo, 16 contiguous k (k = 16h + s, matches B)
      union { uint4 u[2]; v16bf v; } A;
      int k0 = kstep * 32 + 16 * h;
      A.u[0] = *(const uint4*)&wrow[k0];
      A.u[1] = *(const uint4*)&wrow[k0 + 8];
      int base = dy * 66 * 72 + ci0 + 16 * h;
      {   // B fragment, pixel tile n0: col N=nlo, 16 contiguous channels
        const unsigned short* src = &lds[base + (n0 * 16 + nlo + dx) * 72];
        union { uint4 u[2]; v16bf v; } Bm;
        Bm.u[0] = *(const uint4*)&src[0];
        Bm.u[1] = *(const uint4*)&src[8];
        acc0 = __builtin_amdgcn_wmma_f32_16x16x32_bf16(false, A.v, false, Bm.v,
                                                       (short)0, acc0, false, false);
      }
      {   // B fragment, pixel tile n0+1
        const unsigned short* src = &lds[base + ((n0 + 1) * 16 + nlo + dx) * 72];
        union { uint4 u[2]; v16bf v; } Bm;
        Bm.u[0] = *(const uint4*)&src[0];
        Bm.u[1] = *(const uint4*)&src[8];
        acc1 = __builtin_amdgcn_wmma_f32_16x16x32_bf16(false, A.v, false, Bm.v,
                                                       (short)0, acc1, false, false);
      }
    }
  }

  // Epilogue: bias [+BN] [+ReLU], pack 8 bf16 -> one 16B store per tile.
  int coB8 = coBase + m * 16 + 8 * h;
  float bs[8], gg[8], bb2[8];
  for (int r8 = 0; r8 < 8; ++r8) {
    int co = coB8 + r8;
    bs[r8]  = (co < Cout) ? bias[co] : 0.f;
    gg[r8]  = (bng && co < Cout) ? bng[co] : 1.f;
    bb2[r8] = (bng && co < Cout) ? bnb[co] : 0.f;
  }
  for (int t = 0; t < 2; ++t) {
    v8f acc = t ? acc1 : acc0;
    int px = x0 + (n0 + t) * 16 + nlo;
    unsigned short o[8];
    for (int r8 = 0; r8 < 8; ++r8) {
      float vv = acc[r8] + bs[r8];
      if (bng) vv = vv * gg[r8] + bb2[r8];
      if (relu) vv = fmaxf(vv, 0.f);
      if (coB8 + r8 >= Cout) vv = 0.f;
      o[r8] = f2bf(vv);
    }
    *(uint4*)&out[(size_t)((b * HH + y) * WW + px) * CoutPad + coB8] = *(uint4*)o;
  }
}

// ---------------------------------------------------------------------------
// tpn_out: 64 (bf16 channels-last) -> 1, f32, written into xc channel 0.
// ---------------------------------------------------------------------------
__global__ void conv64_out(const unsigned short* __restrict__ in,
                           const float* __restrict__ w, const float* __restrict__ bias,
                           float* __restrict__ xc) {
  int p = blockIdx.x * 256 + threadIdx.x;
  if (p >= PIX) return;
  int x = p % WW; int y = (p / WW) % HH; int b = p / (WW * HH);
  float acc = bias[0];
  for (int dy = 0; dy < 3; ++dy) {
    int gy = y + dy - 1; if ((unsigned)gy >= HH) continue;
    for (int dx = 0; dx < 3; ++dx) {
      int gx = x + dx - 1; if ((unsigned)gx >= WW) continue;
      const unsigned short* ip = &in[(size_t)((b * HH + gy) * WW + gx) * 64];
      int r = dy * 3 + dx;
      for (int ci = 0; ci < 64; ++ci) acc += bf2f(ip[ci]) * w[ci * 9 + r];
    }
  }
  xc[(size_t)p * 2 + 0] = acc;
}

// ---------------------------------------------------------------------------
// Per-pixel softmax(81) x 2 channels + 9x9 dynamic filtering of xc.
// ---------------------------------------------------------------------------
__global__ void dynfilter(const float* __restrict__ xc,
                          const unsigned short* __restrict__ f, // [B,H,W,176] bf16
                          float* __restrict__ out) {
  int p = blockIdx.x * 256 + threadIdx.x;
  if (p >= PIX) return;
  int x = p % WW; int y = (p / WW) % HH; int b = p / (WW * HH);
  const unsigned short* fp = &f[(size_t)p * 176];
  float res = 0.f;
  for (int c = 0; c < 2; ++c) {
    float mx = -1e30f;
    for (int k = 0; k < 81; ++k) mx = fmaxf(mx, bf2f(fp[c * 81 + k]));
    float s = 0.f;
    for (int k = 0; k < 81; ++k) s += __expf(bf2f(fp[c * 81 + k]) - mx);
    float inv = 1.f / s, accum = 0.f;
    for (int i = 0; i < 9; ++i) {
      int gy = y + i - 4;
      for (int j = 0; j < 9; ++j) {
        int gx = x + j - 4;
        float wgt = __expf(bf2f(fp[c * 81 + i * 9 + j]) - mx);
        float pv = ((unsigned)gy < HH && (unsigned)gx < WW)
            ? xc[(size_t)((b * HH + gy) * WW + gx) * 2 + c] : 0.f;
        accum += wgt * pv;
      }
    }
    res += accum * inv;
  }
  out[p] = res;
}

// ---------------------------------------------------------------------------
extern "C" void kernel_launch(void* const* d_in, const int* in_sizes, int n_in,
                              void* d_out, int out_size, void* d_ws, size_t ws_size,
                              hipStream_t stream) {
  (void)in_sizes; (void)n_in; (void)out_size;
  const float* x = (const float*)d_in[0];
  const float* g = (const float*)d_in[1];
  // conv_g layers: [w,b,bn_g,bn_b] starting at index 2, 4 per layer
  const float* cgw[4];  const float* cgb[4];  const float* cgg[4];  const float* cgbb[4];
  for (int i = 0; i < 4; ++i) {
    cgw[i]  = (const float*)d_in[2 + 4 * i];
    cgb[i]  = (const float*)d_in[3 + 4 * i];
    cgg[i]  = (const float*)d_in[4 + 4 * i];
    cgbb[i] = (const float*)d_in[5 + 4 * i];
  }
  const float* tpnin_w  = (const float*)d_in[18];
  const float* tpnin_b  = (const float*)d_in[19];
  const float *tmw[5], *tmb[5], *tmg[5], *tmbb[5];
  for (int i = 0; i < 5; ++i) {
    tmw[i]  = (const float*)d_in[20 + 4 * i];
    tmb[i]  = (const float*)d_in[21 + 4 * i];
    tmg[i]  = (const float*)d_in[22 + 4 * i];
    tmbb[i] = (const float*)d_in[23 + 4 * i];
  }
  const float* tpnout_w = (const float*)d_in[40];
  const float* tpnout_b = (const float*)d_in[41];
  const float* fg0w = (const float*)d_in[42]; const float* fg0b = (const float*)d_in[43];
  const float* fg0g = (const float*)d_in[44]; const float* fg0bb = (const float*)d_in[45];
  const float* fg1w = (const float*)d_in[46]; const float* fg1b = (const float*)d_in[47];
  const float* fg1g = (const float*)d_in[48]; const float* fg1bb = (const float*)d_in[49];
  const float* fow  = (const float*)d_in[50]; const float* fob  = (const float*)d_in[51];

  // ---- workspace carve ----
  char* ws = (char*)d_ws;
  size_t used = 0;
  auto alloc = [&](size_t bytes) -> char* {
    char* p = ws + used; used += (bytes + 255) & ~(size_t)255; return p;
  };
  unsigned short* w2_tpn    = (unsigned short*)alloc((size_t)5 * 64 * 576 * 2);
  unsigned short* w2_fgn2   = (unsigned short*)alloc((size_t)64 * 576 * 2);
  unsigned short* w2_fgnout = (unsigned short*)alloc((size_t)176 * 576 * 2);
  unsigned short* actA = (unsigned short*)alloc((size_t)PIX * 64 * 2);
  unsigned short* actB = (unsigned short*)alloc((size_t)PIX * 64 * 2);
  float* xc = (float*)alloc((size_t)PIX * 2 * 4);
  // union region: g-branch scratch (48MB) first, later reused for fgn_out (176MB)
  char* uni = alloc((size_t)PIX * 176 * 2);
  float* gbufA = (float*)uni;                               // up to [P,16] f32
  float* gbufB = (float*)(uni + (size_t)PIX * 16 * 4);      // up to [P,8] f32
  unsigned short* fgnout = (unsigned short*)uni;            // [P,176] bf16
  if (used > ws_size) return;

  // ---- weight packing ----
  for (int i = 0; i < 5; ++i)
    prep_w2<<<(64 * 576 + 255) / 256, 256, 0, stream>>>(tmw[i], w2_tpn + (size_t)i * 64 * 576, 64, 64);
  prep_w2<<<(64 * 576 + 255) / 256, 256, 0, stream>>>(fg1w, w2_fgn2, 64, 64);
  prep_w2<<<(176 * 576 + 255) / 256, 256, 0, stream>>>(fow, w2_fgnout, 162, 176);

  // ---- conv_g branch (tiny, direct f32) -> xc channel 1 ----
  conv_small<<<(PIX * 16 + 255) / 256, 256, 0, stream>>>(g, cgw[0], cgb[0], cgg[0], cgbb[0],
      gbufA, 1, 16, 1, 16, 0, 1);
  conv_small<<<(PIX * 8 + 255) / 256, 256, 0, stream>>>(gbufA, cgw[1], cgb[1], cgg[1], cgbb[1],
      gbufB, 16, 8, 16, 8, 0, 1);
  conv_small<<<(PIX * 4 + 255) / 256, 256, 0, stream>>>(gbufB, cgw[2], cgb[2], cgg[2], cgbb[2],
      gbufA, 8, 4, 8, 4, 0, 1);
  conv_small<<<(PIX + 255) / 256, 256, 0, stream>>>(gbufA, cgw[3], cgb[3], cgg[3], cgbb[3],
      xc, 4, 1, 4, 2, 1, 2);   // sigmoid, write xc[...,1]

  // ---- TPN branch ----
  conv_in64<<<(PIX * 8 + 255) / 256, 256, 0, stream>>>(x, tpnin_w, tpnin_b,
      nullptr, nullptr, actA, 1, 1);
  dim3 grid64(WW / 64, HH, BB);
  unsigned short* bufs[2] = {actA, actB};
  int cur = 0;
  for (int i = 0; i < 5; ++i) {
    conv64_wmma<<<grid64, 256, 0, stream>>>(bufs[cur], w2_tpn + (size_t)i * 64 * 576,
        tmb[i], tmg[i], tmbb[i], bufs[1 - cur], 64, 64, 1, 1);
    cur = 1 - cur;
  }
  conv64_out<<<(PIX + 255) / 256, 256, 0, stream>>>(bufs[cur], tpnout_w, tpnout_b, xc);

  // ---- filter-generating branch ----
  conv_in64<<<(PIX * 8 + 255) / 256, 256, 0, stream>>>(xc, fg0w, fg0b, fg0g, fg0bb,
      actA, 2, 2);
  conv64_wmma<<<grid64, 256, 0, stream>>>(actA, w2_fgn2, fg1b, fg1g, fg1bb,
      actB, 64, 64, 1, 1);
  dim3 grid176(WW / 64, HH, BB * 3);
  conv64_wmma<<<grid176, 256, 0, stream>>>(actB, w2_fgnout, fob, nullptr, nullptr,
      fgnout, 162, 176, 3, 0);

  // ---- softmax + dynamic filter -> output ----
  dynfilter<<<(PIX + 255) / 256, 256, 0, stream>>>(xc, fgnout, (float*)d_out);
}
